// GCN_48601849921727
// MI455X (gfx1250) — matrix operand order repaired and checked by reading, not compiled, not run on hardware
//
#include <hip/hip_runtime.h>
#include <math.h>

// GCN forward for MI455X (gfx1250), wave32, WMMA bf16 16x16x32.
// Weights pre-converted once to bf16 transposed layout (prep kernels), so
// the GEMM hot loops stage pure-bf16 slabs (copy only, no conversion).
// Pipeline: prep(W1T,W2T) -> gemm1(WMMA) -> spmm1 -> bias/relu ->
//           gemm2(WMMA) -> spmm2 -> bias + log_softmax.

typedef __attribute__((ext_vector_type(16))) __bf16 v16bf;
typedef __attribute__((ext_vector_type(2)))  __bf16 v2bf;
typedef __attribute__((ext_vector_type(16))) float  v16f;
typedef __attribute__((ext_vector_type(2)))  float  v2f;
typedef __attribute__((ext_vector_type(8)))  float  v8f;

#define F_IN 512
#define HID  128
#define NCLS 40

__device__ __forceinline__ void fatomic_add(float* p, float v) {
  __hip_atomic_fetch_add(p, v, __ATOMIC_RELAXED, __HIP_MEMORY_SCOPE_AGENT);
}

__device__ __forceinline__ v8f zero8() {
  v8f z = {0.f, 0.f, 0.f, 0.f, 0.f, 0.f, 0.f, 0.f};
  return z;
}

// Pack two floats into adjacent bf16 (one v_cvt_pk_bf16_f32) and store 4B.
__device__ __forceinline__ void store_pk(__bf16* dst, float lo, float hi) {
  v2f t = {lo, hi};
  *(v2bf*)dst = __builtin_convertvector(t, v2bf);
}

// Load A fragment (16x32, contiguous k = 16*half + s, same order as B): row fixed per lane,
// K = abase + {0..15}: one contiguous 16-float run -> 4x b128 loads,
// one vector convert (lowers to v_cvt_pk_bf16_f32 pairs).
__device__ __forceinline__ v16bf load_a_frag(const float* __restrict__ rowp,
                                             int abase) {
  const float* xp = rowp + abase;
  float4 f0 = *(const float4*)(xp);
  float4 f1 = *(const float4*)(xp + 4);
  float4 f2 = *(const float4*)(xp + 8);
  float4 f3 = *(const float4*)(xp + 12);
  v16f af;
  af[0]  = f0.x; af[1]  = f0.y; af[2]  = f0.z; af[3]  = f0.w;
  af[4]  = f1.x; af[5]  = f1.y; af[6]  = f1.z; af[7]  = f1.w;
  af[8]  = f2.x; af[9]  = f2.y; af[10] = f2.z; af[11] = f2.w;
  af[12] = f3.x; af[13] = f3.y; af[14] = f3.z; af[15] = f3.w;
  return __builtin_convertvector(af, v16bf);
}

// ---------------------------------------------------------------------------
// Prep: W1 [512,128] f32 -> W1T [n=128][k=512] bf16 (k-pair packed stores).
// 32768 threads, coalesced reads.
// ---------------------------------------------------------------------------
__global__ __launch_bounds__(256) void prep_w1t_kernel(
    const float* __restrict__ W1, __bf16* __restrict__ W1T) {
  int t  = blockIdx.x * 256 + threadIdx.x;       // 0..32767
  int n  = t & 127;
  int k  = (t >> 7) * 2;                         // 0,2,..,510
  float a0 = W1[(size_t)(k + 0) * HID + n];
  float a1 = W1[(size_t)(k + 1) * HID + n];
  store_pk(&W1T[(size_t)n * F_IN + k], a0, a1);
}

// Prep: W2 [128,40] f32 -> W2T [n=48][k=128] bf16, zero-padded cols 40..47.
__global__ __launch_bounds__(256) void prep_w2t_kernel(
    const float* __restrict__ W2, __bf16* __restrict__ W2T) {
  int t = blockIdx.x * 256 + threadIdx.x;        // 0..3071
  if (t >= 48 * 64) return;
  int n = t % 48;
  int k = (t / 48) * 2;
  float a0 = (n < NCLS) ? W2[(k + 0) * NCLS + n] : 0.f;
  float a1 = (n < NCLS) ? W2[(k + 1) * NCLS + n] : 0.f;
  store_pk(&W2T[(size_t)n * HID + k], a0, a1);
}

// ---------------------------------------------------------------------------
// GEMM1: h0[N,128] = x[N,512] @ W1[512,128]  (bf16 WMMA, f32 accumulate)
// Block = 128 rows (8 waves x 16 rows); each wave owns all 8 N-tiles.
// W1T slabs copied (bf16, no conversion) into double-buffered LDS [n][k]:
// one 32B chunk per thread per K-step. B fragments ping-ponged so ds_load
// latency overlaps WMMA issue.
// ---------------------------------------------------------------------------
__global__ __launch_bounds__(256) void gemm1_kernel(
    const float* __restrict__ x, const __bf16* __restrict__ W1T,
    float* __restrict__ h0, int N) {
  __shared__ __attribute__((aligned(32))) __bf16 sBT[2][HID * 32];  // 16 KB

  const int tid   = threadIdx.x;
  const int wave  = tid >> 5;
  const int lane  = tid & 31;
  const int m0    = blockIdx.x * 128 + wave * 16;
  const int mrow  = m0 + (lane & 15);
  const int rowc  = (mrow < N) ? mrow : (N - 1);   // clamp; OOB rows not stored
  const int abase = (lane < 16) ? 0 : 16;          // A frag K base (contiguous)
  const int bbase = (lane < 16) ? 0 : 16;          // B frag K base
  const int col   = lane & 15;

  // Staging: thread copies one 32B chunk (n = tid>>1, 16 bf16 at half*16).
  const int c_n = tid >> 1;
  const int c_h = (tid & 1) * 16;
  const __bf16* wsrc = W1T + (size_t)c_n * F_IN + c_h;
  __bf16*       wdst0 = &sBT[0][c_n * 32 + c_h];
  __bf16*       wdst1 = &sBT[1][c_n * 32 + c_h];

  v8f acc[8];
#pragma unroll
  for (int i = 0; i < 8; ++i) acc[i] = zero8();

  const float* rowp = x + (size_t)rowc * F_IN;

  // Prologue: copy slab kt=0.
  *(v16bf*)wdst0 = *(const v16bf*)(wsrc);
  __syncthreads();

  for (int kt = 0; kt < F_IN / 32; ++kt) {
    const int cur = kt & 1;
    const bool have_next = (kt + 1) < (F_IN / 32);   // uniform

    // (a1) issue next slab's global loads early (overlap with WMMA)
    v16bf stg;
    if (have_next) stg = *(const v16bf*)(wsrc + (kt + 1) * 32);

    // (b) compute: A frag + ping-ponged B frags + WMMAs
    v16bf a = load_a_frag(rowp + kt * 32, abase);
    const __bf16* bbasep = &sBT[cur][col * 32 + bbase];
    v16bf bc = *(const v16bf*)(bbasep);
#pragma unroll
    for (int nt = 0; nt < 8; ++nt) {
      v16bf bn;
      if (nt < 7) bn = *(const v16bf*)(bbasep + (nt + 1) * 16 * 32);
      acc[nt] = __builtin_amdgcn_wmma_f32_16x16x32_bf16(
          false, a, false, bc, (short)0, acc[nt], false, false);
      bc = bn;
    }

    // (a2) store staged slab into the other buffer
    if (have_next) *(v16bf*)((cur == 0) ? wdst1 : wdst0) = stg;
    __syncthreads();
  }

  // C/D layout: VGPR r -> M = (lane<16?0:8)+r, N = lane%16
  const int moff = (lane < 16) ? 0 : 8;
  float* op = h0 + (size_t)(m0 + moff) * HID + col;
  if (m0 + 16 <= N) {                              // wave-uniform fast path
#pragma unroll
    for (int nt = 0; nt < 8; ++nt)
#pragma unroll
      for (int r = 0; r < 8; ++r)
        op[(size_t)r * HID + nt * 16] = acc[nt][r];
  } else {                                         // tail block only
#pragma unroll
    for (int nt = 0; nt < 8; ++nt)
#pragma unroll
      for (int r = 0; r < 8; ++r)
        if (m0 + moff + r < N) op[(size_t)r * HID + nt * 16] = acc[nt][r];
  }
}

// ---------------------------------------------------------------------------
// GEMM2: l0[N,40] = h[N,128] @ W2[128,40]  (bf16 WMMA, N padded 40 -> 48)
// W2T copied once into LDS (pure bf16 memcpy, 3x b128 per thread).
// ---------------------------------------------------------------------------
__global__ __launch_bounds__(256) void gemm2_kernel(
    const float* __restrict__ h, const __bf16* __restrict__ W2T,
    float* __restrict__ l0, int N) {
  __shared__ __attribute__((aligned(32))) __bf16 sBT[48 * HID];  // [n][k], 12 KB

  const int tid = threadIdx.x;
#pragma unroll
  for (int j = 0; j < 3; ++j) {
    int idx = tid + j * 256;                       // 768 chunks of 16B
    ((uint4*)sBT)[idx] = ((const uint4*)W2T)[idx];
  }
  __syncthreads();

  const int wave  = tid >> 5;
  const int lane  = tid & 31;
  const int m0    = blockIdx.x * 128 + wave * 16;
  const int mrow  = m0 + (lane & 15);
  const int rowc  = (mrow < N) ? mrow : (N - 1);
  const int abase = (lane < 16) ? 0 : 16;
  const int bbase = (lane < 16) ? 0 : 16;
  const int col   = lane & 15;

  v8f acc[3];
#pragma unroll
  for (int i = 0; i < 3; ++i) acc[i] = zero8();

  const float* rowp = h + (size_t)rowc * HID;

#pragma unroll
  for (int kt = 0; kt < HID / 32; ++kt) {
    v16bf a = load_a_frag(rowp + kt * 32, abase);
    v16bf b[3];
#pragma unroll
    for (int nt = 0; nt < 3; ++nt)
      b[nt] = *(const v16bf*)(&sBT[(nt * 16 + col) * HID + kt * 32 + bbase]);
#pragma unroll
    for (int nt = 0; nt < 3; ++nt)
      acc[nt] = __builtin_amdgcn_wmma_f32_16x16x32_bf16(
          false, a, false, b[nt], (short)0, acc[nt], false, false);
  }

  const int moff = (lane < 16) ? 0 : 8;
  float* op = l0 + (size_t)(m0 + moff) * NCLS + col;
  if (m0 + 16 <= N) {                              // wave-uniform fast path
#pragma unroll
    for (int nt = 0; nt < 2; ++nt)                 // cols 0..31 always valid
#pragma unroll
      for (int r = 0; r < 8; ++r)
        op[(size_t)r * NCLS + nt * 16] = acc[nt][r];
    if (col < 8) {                                 // cols 32..39 of tile 2
#pragma unroll
      for (int r = 0; r < 8; ++r)
        op[(size_t)r * NCLS + 32] = acc[2][r];
    }
  } else {
#pragma unroll
    for (int nt = 0; nt < 3; ++nt) {
      int n = nt * 16 + col;
      if (n < NCLS) {
#pragma unroll
        for (int r = 0; r < 8; ++r)
          if (m0 + moff + r < N) op[(size_t)r * NCLS + nt * 16] = acc[nt][r];
      }
    }
  }
}

// ---------------------------------------------------------------------------
// SpMM1: h[dst,:] += w * h0[src,:]  (d=128, 32 lanes/edge, float4 per lane)
// ---------------------------------------------------------------------------
__global__ __launch_bounds__(256) void spmm1_kernel(
    const long long* __restrict__ ei, const float* __restrict__ ew,
    const float* __restrict__ h0, float* __restrict__ h, int E) {
  int t = blockIdx.x * 256 + threadIdx.x;
  int e = t >> 5;
  if (e >= E) return;
  int g   = (t & 31) * 4;
  int dst = (int)ei[e];
  int src = (int)ei[(size_t)E + e];
  float w = ew[e];
  const float4 v = *(const float4*)(h0 + (size_t)src * HID + g);
  float* o = h + (size_t)dst * HID + g;
  fatomic_add(o + 0, w * v.x);
  fatomic_add(o + 1, w * v.y);
  fatomic_add(o + 2, w * v.z);
  fatomic_add(o + 3, w * v.w);
}

// ---------------------------------------------------------------------------
// SpMM2: out[dst,f] += w * l0[src,f]  (d=40, 32 lanes/edge; lane handles
// f=lane, and f=lane+32 for lanes 0..7 -- no integer division)
// ---------------------------------------------------------------------------
__global__ __launch_bounds__(256) void spmm2_kernel(
    const long long* __restrict__ ei, const float* __restrict__ ew,
    const float* __restrict__ l0, float* __restrict__ out, int E) {
  int t = blockIdx.x * 256 + threadIdx.x;
  int e = t >> 5;
  if (e >= E) return;
  int lane = t & 31;
  int dst = (int)ei[e];
  int src = (int)ei[(size_t)E + e];
  float w = ew[e];
  const float* sp = l0 + (size_t)src * NCLS;
  float*       op = out + (size_t)dst * NCLS;
  fatomic_add(op + lane, w * sp[lane]);
  if (lane < (NCLS - 32))
    fatomic_add(op + lane + 32, w * sp[lane + 32]);
}

// ---------------------------------------------------------------------------
// Elementwise helpers (float4 vectorized)
// ---------------------------------------------------------------------------
__global__ __launch_bounds__(256) void zero4_kernel(float4* __restrict__ p, int n4) {
  int t = blockIdx.x * 256 + threadIdx.x;
  if (t < n4) p[t] = float4{0.f, 0.f, 0.f, 0.f};
}

__global__ __launch_bounds__(256) void bias_relu_kernel(
    float* __restrict__ h, const float* __restrict__ b1, int n4) {
  int t = blockIdx.x * 256 + threadIdx.x;
  if (t < n4) {
    float4 v = *(const float4*)(h + (size_t)t * 4);
    float4 b = *(const float4*)(b1 + ((t * 4) & (HID - 1)));
    v.x = fmaxf(v.x + b.x, 0.f);
    v.y = fmaxf(v.y + b.y, 0.f);
    v.z = fmaxf(v.z + b.z, 0.f);
    v.w = fmaxf(v.w + b.w, 0.f);
    *(float4*)(h + (size_t)t * 4) = v;
  }
}

// ---------------------------------------------------------------------------
// Fused bias + log_softmax over 40 classes. One wave32 per row.
// ---------------------------------------------------------------------------
__global__ __launch_bounds__(256) void lsm_kernel(
    float* __restrict__ out, const float* __restrict__ b2, int N) {
  int row  = blockIdx.x * 8 + (threadIdx.x >> 5);
  int lane = threadIdx.x & 31;
  if (row >= N) return;  // wave-uniform branch
  float* p = out + (size_t)row * NCLS;

  float va   = p[lane] + b2[lane];
  bool  has2 = lane < (NCLS - 32);
  float vb   = has2 ? (p[lane + 32] + b2[lane + 32]) : 0.f;

  float m = has2 ? fmaxf(va, vb) : va;
#pragma unroll
  for (int off = 16; off > 0; off >>= 1)
    m = fmaxf(m, __shfl_xor(m, off, 32));

  float s = __expf(va - m) + (has2 ? __expf(vb - m) : 0.f);
#pragma unroll
  for (int off = 16; off > 0; off >>= 1)
    s += __shfl_xor(s, off, 32);

  float lse = m + __logf(s);
  p[lane] = va - lse;
  if (has2) p[lane + 32] = vb - lse;
}

// ---------------------------------------------------------------------------
// Launcher
// ---------------------------------------------------------------------------
extern "C" void kernel_launch(void* const* d_in, const int* in_sizes, int n_in,
                              void* d_out, int out_size, void* d_ws, size_t ws_size,
                              hipStream_t stream) {
  const float*     x  = (const float*)d_in[0];
  const long long* ei = (const long long*)d_in[1];  // int64 [2,E]
  const float*     ew = (const float*)d_in[2];
  const float*     W1 = (const float*)d_in[3];
  const float*     b1 = (const float*)d_in[4];
  const float*     W2 = (const float*)d_in[5];
  const float*     b2 = (const float*)d_in[6];
  float*           out = (float*)d_out;

  const int N = in_sizes[0] / F_IN;   // 100000
  const int E = in_sizes[2];          // 1600000

  // Workspace: h0 [N*128] f32 | h [N*128] f32 | W1T bf16 | W2T bf16.
  // l0 aliases h0 (dead after spmm1).
  float*  h0  = (float*)d_ws;
  float*  h   = h0 + (size_t)N * HID;
  float*  l0  = h0;
  __bf16* W1T = (__bf16*)(h + (size_t)N * HID);
  __bf16* W2T = W1T + (size_t)F_IN * HID;

  const int mb  = (N + 127) / 128;
  const int nh4 = N * HID / 4;        // divisible (HID = 128)
  const int no4 = N * NCLS / 4;       // divisible (NCLS = 40)

  prep_w1t_kernel<<<(F_IN * HID / 2) / 256, 256, 0, stream>>>(W1, W1T);
  prep_w2t_kernel<<<(48 * 64 + 255) / 256, 256, 0, stream>>>(W2, W2T);
  zero4_kernel<<<(nh4 + 255) / 256, 256, 0, stream>>>((float4*)h, nh4);
  gemm1_kernel<<<mb, 256, 0, stream>>>(x, W1T, h0, N);
  spmm1_kernel<<<((size_t)E * 32 + 255) / 256, 256, 0, stream>>>(ei, ew, h0, h, E);
  bias_relu_kernel<<<(nh4 + 255) / 256, 256, 0, stream>>>(h, b1, nh4);
  gemm2_kernel<<<mb, 256, 0, stream>>>(h, W2T, l0, N);
  zero4_kernel<<<(no4 + 255) / 256, 256, 0, stream>>>((float4*)out, no4);
  spmm2_kernel<<<((size_t)E * 32 + 255) / 256, 256, 0, stream>>>(ei, ew, l0, out, E);
  lsm_kernel<<<(N + 7) / 8, 256, 0, stream>>>(out, b2, N);
}
